// AutoregressiveFlow_75591424409755
// MI455X (gfx1250) — matrix operand order repaired and checked by reading, not compiled, not run on hardware
//
#include <hip/hip_runtime.h>
#include <stdint.h>

#define B_TOT   8192
#define L_SEQ   256
#define S_STEPS 255
#define H       512
#define NOUT    48
#define BT      64
#define TWO_PI_F      6.283185307179586f
#define INV_TWO_PI_F  0.15915494309189535f
#define EPS_F         1e-9f

typedef __bf16 v16bf __attribute__((ext_vector_type(16)));
typedef float  v8f   __attribute__((ext_vector_type(8)));
typedef unsigned int u32x4v __attribute__((ext_vector_type(4)));
typedef int          i32x8v __attribute__((ext_vector_type(8)));
typedef int          i32x4v __attribute__((ext_vector_type(4)));

union FragBF { uint4 q[2]; v16bf v; };

#if defined(__has_builtin)
#if __has_builtin(__builtin_amdgcn_tensor_load_to_lds) && __has_builtin(__builtin_amdgcn_s_wait_tensorcnt)
#define HAS_TDM 1
#endif
#endif
#ifndef HAS_TDM
#define HAS_TDM 0
#endif

__device__ __forceinline__ unsigned short f2bf(float f) {
  unsigned u = __float_as_uint(f);
  u += 0x7FFFu + ((u >> 16) & 1u);          // round-to-nearest-even
  return (unsigned short)(u >> 16);
}

__device__ __forceinline__ float mod2pi(float x) {
  float r = fmodf(x, TWO_PI_F);
  return r < 0.f ? r + TWO_PI_F : r;
}

// ramp(x) = exp(-x^-2) for x>0 (x clipped to >=1e-6), else 0   (P = 2)
__device__ __forceinline__ float ramp_f(float x) {
  float xs = fmaxf(x, 1e-6f);
  float r  = __expf(-1.0f / (xs * xs));
  return x > 0.f ? r : 0.f;
}
__device__ __forceinline__ float dramp_f(float x) {
  float xs   = fmaxf(x, 1e-6f);
  float inv  = 1.0f / xs;
  float inv2 = inv * inv;
  float r    = __expf(-inv2) * 2.0f * inv2 * inv;   // exp(-x^-2) * 2 * x^-3
  return x > 0.f ? r : 0.f;
}
__device__ __forceinline__ float sig_f(float zv) {
  float r1 = ramp_f(zv), r2 = ramp_f(1.f - zv);
  return r1 / (r1 + r2 + EPS_F);
}
__device__ __forceinline__ float dsig_f(float zv) {
  float r1 = ramp_f(zv), r2 = ramp_f(1.f - zv);
  float d1 = dramp_f(zv), d2 = dramp_f(1.f - zv);
  float s  = r1 + r2;
  return (d1 * r2 + r1 * d2) / (s * s + EPS_F);
}

#if HAS_TDM
// 2-D TDM load: tile [tile_n rows][tile_k elems] of bf16 from a row-major
// [td1][stride] tensor into contiguous LDS at lds_off.
__device__ __forceinline__ void tdm_load_2d(unsigned lds_off, const void* gptr,
                                            unsigned tile_k, unsigned tile_n,
                                            unsigned stride_elems,
                                            unsigned td0, unsigned td1) {
  unsigned long long ga = (unsigned long long)(uintptr_t)gptr;
  u32x4v g0;
  g0.x = 1u;                                              // count=1, user mode
  g0.y = lds_off;                                         // LDS byte offset
  g0.z = (unsigned)(ga & 0xFFFFFFFFull);                  // global_addr[31:0]
  g0.w = ((unsigned)(ga >> 32) & 0x01FFFFFFu) | (2u << 30); // addr[56:32] | type=2
  i32x8v g1;
  g1[0] = (int)(1u << 16);                                // data_size = 2 bytes
  g1[1] = (int)((td0 & 0xFFFFu) << 16);                   // tensor_dim0[15:0]
  g1[2] = (int)(((td0 >> 16) & 0xFFFFu) | ((td1 & 0xFFFFu) << 16));
  g1[3] = (int)(((td1 >> 16) & 0xFFFFu) | (tile_k << 16));// tile_dim0
  g1[4] = (int)(tile_n & 0xFFFFu);                        // tile_dim1 (tile_dim2=0)
  g1[5] = (int)stride_elems;                              // tensor_dim0_stride[31:0]
  g1[6] = 0;
  g1[7] = 0;
  i32x4v z4 = {0, 0, 0, 0};
  i32x8v z8 = {0, 0, 0, 0, 0, 0, 0, 0};
  // 6-arg form (therock/clang-23 lane): groups 0..3 + extra group + cpol
  __builtin_amdgcn_tensor_load_to_lds(g0, g1, z4, z4, z8, 0);
}
#endif

// ---------------- one-time weight conversion: fp32 [S][K][N] -> bf16 [S][N][K]
__global__ void convert_w1_k(const float* __restrict__ W1,
                             unsigned short* __restrict__ W1bf) {
  const size_t total = (size_t)S_STEPS * H * H;
  for (size_t i = (size_t)blockIdx.x * blockDim.x + threadIdx.x; i < total;
       i += (size_t)gridDim.x * blockDim.x) {
    size_t st = i / ((size_t)H * H);
    size_t r  = i - st * H * H;
    size_t n  = r / H, k = r - n * H;
    W1bf[i] = f2bf(W1[st * H * H + k * H + n]);
  }
}
__global__ void convert_w2_k(const float* __restrict__ W2,
                             unsigned short* __restrict__ W2bf) {
  const size_t total = (size_t)S_STEPS * NOUT * H;
  for (size_t i = (size_t)blockIdx.x * blockDim.x + threadIdx.x; i < total;
       i += (size_t)gridDim.x * blockDim.x) {
    size_t st = i / ((size_t)NOUT * H);
    size_t r  = i - st * NOUT * H;
    size_t n  = r / H, k = r - n * H;
    W2bf[i] = f2bf(W2[st * H * NOUT + k * NOUT + n]);
  }
}

// ---------------- main persistent flow kernel: 64 batch rows per workgroup
__global__ __launch_bounds__(256, 1)
void flow_main(const float* __restrict__ z,  const float* __restrict__ W0,
               const float* __restrict__ b0, const float* __restrict__ b1,
               const float* __restrict__ b2,
               const unsigned short* __restrict__ W1bf,
               const unsigned short* __restrict__ W2bf,
               float* __restrict__ out) {
  __shared__ __align__(16) unsigned short sA [BT * H];    // layer0 act (bf16)
  __shared__ __align__(16) unsigned short sA2[BT * H];    // layer1 act (bf16)
  __shared__ __align__(16) unsigned short sB [H * 64];    // W1 K-chunk [N=512][Kc=64]
  __shared__ __align__(16) unsigned short sW2[NOUT * H];  // W2 step tile [48][512]
  __shared__ __align__(16) float sRaw[BT * NOUT];
  __shared__ float sW0[3 * H];                            // W0 row0|row1|b0
  __shared__ float sPhi[BT], sCos[BT], sSin[BT];

  const int tid   = threadIdx.x;
  const int wave  = tid >> 5;
  const int lane  = tid & 31;
  const int lhalf = (lane >> 4) & 1;
  const int l16   = lane & 15;
  const int rowBase = blockIdx.x * BT;

  float phi0_r = 0.f, ldj_r = 0.f;
  if (tid < BT) {
    int b = rowBase + tid;
    float p0 = z[(size_t)b * L_SEQ];
    sPhi[tid] = p0; sCos[tid] = 1.f; sSin[tid] = 0.f;     // angle = 0 initially
    out[(size_t)b * L_SEQ] = p0;
    phi0_r = p0;
  }
  __syncthreads();

  for (int s = 0; s < S_STEPS; ++s) {
    const unsigned short* w2p = W2bf + (size_t)s * NOUT * H;
    const unsigned short* w1p = W1bf + (size_t)s * H * H;

#if HAS_TDM
    if (tid < 32) {           // one TDM issue per workgroup (wave 0)
      tdm_load_2d((unsigned)(uintptr_t)sW2, w2p, /*tile_k=*/H, /*tile_n=*/NOUT,
                  /*stride=*/H, /*td0=*/H, /*td1=*/NOUT);
    }
#else
    for (int q = tid; q < NOUT * H / 8; q += 256)
      ((uint4*)sW2)[q] = ((const uint4*)w2p)[q];
#endif
    // stage W0/b0 for this step
    for (int i = tid; i < 3 * H; i += 256)
      sW0[i] = (i < 2 * H) ? W0[(size_t)s * 2 * H + i]
                           : b0[(size_t)s * H + (i - 2 * H)];
    __syncthreads();

    // ---- layer 0: ctx(2) -> H, tanh, bf16 into sA
    for (int idx = tid; idx < BT * H; idx += 256) {
      int r = idx >> 9, h = idx & (H - 1);
      float v = tanhf(fmaf(sCos[r], sW0[h], fmaf(sSin[r], sW0[H + h], sW0[2 * H + h])));
      sA[idx] = f2bf(v);
    }

    // ---- layer 1: (64x512)@(512x512) via bf16 WMMA, K chunked by 64
    v8f acc[4][4];
    #pragma unroll
    for (int mt = 0; mt < 4; ++mt)
      #pragma unroll
      for (int nl = 0; nl < 4; ++nl)
        acc[mt][nl] = (v8f){0.f, 0.f, 0.f, 0.f, 0.f, 0.f, 0.f, 0.f};

    for (int kb = 0; kb < 8; ++kb) {
#if HAS_TDM
      if (tid < 32) {
        tdm_load_2d((unsigned)(uintptr_t)sB, w1p + kb * 64, /*tile_k=*/64,
                    /*tile_n=*/H, /*stride=*/H, /*td0=*/H, /*td1=*/H);
        __builtin_amdgcn_s_wait_tensorcnt(0);   // also covers the W2 TDM
      }
      __syncthreads();
#else
      __syncthreads();
      for (int q = tid; q < H * 8; q += 256) {  // 512 rows x 64 elems = 4096 uint4
        int rr = q >> 3, c = (q & 7) * 8;
        *(uint4*)&sB[rr * 64 + c] = *(const uint4*)&w1p[(size_t)rr * H + kb * 64 + c];
      }
      __syncthreads();
#endif
      #pragma unroll
      for (int k2 = 0; k2 < 2; ++k2) {
        const int kc32 = k2 * 32;
        FragBF bf[4];
        #pragma unroll
        for (int nl = 0; nl < 4; ++nl) {        // B: lane-half gets contiguous K+0/K+16
          int n = (wave * 4 + nl) * 16 + l16;
          const unsigned short* p = &sB[n * 64 + kc32 + lhalf * 16];
          bf[nl].q[0] = *(const uint4*)p;
          bf[nl].q[1] = *(const uint4*)(p + 8);
        }
        #pragma unroll
        for (int mt = 0; mt < 4; ++mt) {        // A: lane-half gets contiguous K+0/K+16
          FragBF af;
          int m = mt * 16 + l16;
          const unsigned short* p = &sA[m * H + kb * 64 + kc32 + lhalf * 16];
          af.q[0] = *(const uint4*)p;
          af.q[1] = *(const uint4*)(p + 8);
          #pragma unroll
          for (int nl = 0; nl < 4; ++nl)
            acc[mt][nl] = __builtin_amdgcn_wmma_f32_16x16x32_bf16(
                false, af.v, false, bf[nl].v, (short)0, acc[mt][nl], false, false);
        }
      }
      __syncthreads();
    }

    // bias + tanh -> sA2 (bf16).  C layout: lane l, vgpr r -> M = 8*(l>=16)+r, N = l%16
    #pragma unroll
    for (int mt = 0; mt < 4; ++mt)
      #pragma unroll
      for (int nl = 0; nl < 4; ++nl) {
        int n = (wave * 4 + nl) * 16 + l16;
        float bias = b1[(size_t)s * H + n];
        int m0 = mt * 16 + lhalf * 8;
        #pragma unroll
        for (int r = 0; r < 8; ++r)
          sA2[(m0 + r) * H + n] = f2bf(tanhf(acc[mt][nl][r] + bias));
      }
    __syncthreads();

    // ---- layer 2: (64x512)@(512x48): 12 tiles spread over 8 waves
    #pragma unroll
    for (int ti = 0; ti < 2; ++ti) {
      int t = wave + ti * 8;
      if (t < 12) {
        int mt = t / 3, nt = t % 3;
        v8f a2 = (v8f){0.f, 0.f, 0.f, 0.f, 0.f, 0.f, 0.f, 0.f};
        #pragma unroll
        for (int k32 = 0; k32 < 16; ++k32) {
          FragBF af, bf;
          int m = mt * 16 + l16;
          const unsigned short* pa = &sA2[m * H + k32 * 32 + lhalf * 16];
          af.q[0] = *(const uint4*)pa;
          af.q[1] = *(const uint4*)(pa + 8);
          int n = nt * 16 + l16;
          const unsigned short* pb = &sW2[n * H + k32 * 32 + lhalf * 16];
          bf.q[0] = *(const uint4*)pb;
          bf.q[1] = *(const uint4*)(pb + 8);
          a2 = __builtin_amdgcn_wmma_f32_16x16x32_bf16(
              false, af.v, false, bf.v, (short)0, a2, false, false);
        }
        int n = nt * 16 + l16;
        float bias = b2[(size_t)s * NOUT + n];
        int m0 = mt * 16 + lhalf * 8;
        #pragma unroll
        for (int r = 0; r < 8; ++r)
          sRaw[(m0 + r) * NOUT + n] = a2[r] + bias;
      }
    }
    __syncthreads();

    // ---- mixture transform + phi/ldj update (one thread per row)
    if (tid < BT) {
      int row = tid, b = rowBase + tid;
      float phi_prev = sPhi[row];
      float Vv = mod2pi(z[(size_t)b * L_SEQ + (s + 1)] - phi_prev);
      float x  = Vv * INV_TWO_PI_F;
      const float* raw = &sRaw[row * NOUT];
      float wmax = raw[32];
      #pragma unroll
      for (int j = 1; j < 16; ++j) wmax = fmaxf(wmax, raw[32 + j]);
      float we[16], wsum = 0.f;
      #pragma unroll
      for (int j = 0; j < 16; ++j) { we[j] = __expf(raw[32 + j] - wmax); wsum += we[j]; }
      float wscale = (1.0f - 16.0f * 0.01f) / wsum;
      float Hx = 0.f, dHx = 0.f;
      #pragma unroll
      for (int j = 0; j < 16; ++j) {
        float ar = raw[j];
        float a  = 1.f + (ar > 20.f ? ar : log1pf(__expf(ar)));   // 1 + softplus
        float bb = 1.f / (1.f + __expf(-raw[16 + j]));            // sigmoid
        float w  = we[j] * wscale + 0.01f;
        float zv = a * (x - bb);
        float s0 = sig_f(-a * bb);
        float s1 = sig_f(a * (1.f - bb));
        float norm = s1 - s0 + EPS_F;
        Hx  += w * ((sig_f(zv) - s0) / norm);
        dHx += w * (a * dsig_f(zv) / norm);
      }
      float U = TWO_PI_F * Hx;
      ldj_r += __logf(dHx + EPS_F);
      float phi_new = mod2pi(U + phi_prev);
      out[(size_t)b * L_SEQ + (s + 1)] = phi_new;
      sPhi[row] = phi_new;
      float ang = phi_new - phi0_r;
      sCos[row] = __cosf(ang);
      sSin[row] = __sinf(ang);
    }
    __syncthreads();
  }

  if (tid < BT) out[(size_t)B_TOT * L_SEQ + rowBase + tid] = ldj_r;
}

extern "C" void kernel_launch(void* const* d_in, const int* in_sizes, int n_in,
                              void* d_out, int out_size, void* d_ws, size_t ws_size,
                              hipStream_t stream) {
  const float* z  = (const float*)d_in[0];
  const float* W0 = (const float*)d_in[1];
  const float* b0 = (const float*)d_in[2];
  const float* W1 = (const float*)d_in[3];
  const float* b1 = (const float*)d_in[4];
  const float* W2 = (const float*)d_in[5];
  const float* b2 = (const float*)d_in[6];
  float* out = (float*)d_out;

  unsigned short* W1bf = (unsigned short*)d_ws;                       // 255*512*512 bf16
  unsigned short* W2bf = W1bf + (size_t)S_STEPS * H * H;              // 255*48*512 bf16

  convert_w1_k<<<2048, 256, 0, stream>>>(W1, W1bf);
  convert_w2_k<<<512, 256, 0, stream>>>(W2, W2bf);
  flow_main<<<B_TOT / BT, 256, 0, stream>>>(z, W0, b0, b1, b2, W1bf, W2bf, out);
}
